// GraphSAGELSTMBaseline_45354854646279
// MI455X (gfx1250) — matrix operand order repaired and checked by reading, not compiled, not run on hardware
//
#include <hip/hip_runtime.h>
#include <hip/hip_bf16.h>

// ---------------------------------------------------------------------------
// GraphSAGE + LSTM for MI455X (gfx1250), wave32 + WMMA bf16.
//
//   kA zero   : clear CSR counts + cursors
//   kB count  : per-edge in-degree histogram (u32 atomics, 800k not 51.2M)
//   kC scan   : single-block exclusive scan -> CSR row offsets
//   kD fill   : scatter src ids into CSR adjacency
//   k2 pack   : transpose+convert all weight matrices to bf16 [N][K]
//   kE asage  : wave-per-node gather-mean + fused A_sage=bf16([x|mean]) build
//   k4 sage   : h_struct = A_sage @ [W_self;W_neigh] + b   (WMMA) [N,128] f32
//   k5 lstm   : fused 8-step LSTM, weights resident in 256KB LDS (WMMA)
//   k6 cls    : out = bf16(h_struct + h_T) @ W_cls + b     (WMMA)
//
// Row-indexed scratch matrices padded to NPAD = 50048 rows => branch-free
// GEMM prologues/epilogues; only the final d_out store is guarded (uniform
// branch, slow path in 1 of 391 blocks). All accumulators zero-init so the
// first v_wmma of each chain takes the inline-0 C operand (no splat movs);
// biases are added in the epilogue where they overlap the XDL pipe.
// ---------------------------------------------------------------------------

#define NN      50000
#define NPAD    50048           // 391 * 128
#define NE      800000
#define INDIM   64
#define HID     128
#define OUTDIM  64
#define THIST   8

typedef __attribute__((ext_vector_type(16))) __bf16 bf16x16;
typedef __attribute__((ext_vector_type(8)))  __bf16 bf16x8;
typedef __attribute__((ext_vector_type(8)))  float  f32x8;

__device__ __forceinline__ bf16x16 combine16(bf16x8 lo, bf16x8 hi) {
  return __builtin_shufflevector(lo, hi, 0,1,2,3,4,5,6,7,8,9,10,11,12,13,14,15);
}
__device__ __forceinline__ bf16x16 zfrag() {
  bf16x16 v;
#pragma unroll
  for (int i = 0; i < 16; ++i) v[i] = (__bf16)0.0f;
  return v;
}
__device__ __forceinline__ f32x8 zacc() {
  f32x8 v;
#pragma unroll
  for (int i = 0; i < 8; ++i) v[i] = 0.0f;
  return v;
}
// A-fragment (16x32 bf16): lane holds row lane%16; K = {0..7,16..23} (lanes<16)
// or {8..15,24..31} (lanes>=16). `p` already includes row*ld + k0 + khA.
__device__ __forceinline__ bf16x16 afrag_bf16(const __bf16* p) {
  bf16x8 lo = *(const bf16x8*)(p);
  bf16x8 hi = *(const bf16x8*)(p + 16);
  return combine16(lo, hi);
}
__device__ __forceinline__ bf16x16 afrag_f32(const float* p) {
  f32x8 lo = *(const f32x8*)(p);
  f32x8 hi = *(const f32x8*)(p + 16);
  bf16x16 r;
#pragma unroll
  for (int i = 0; i < 8; ++i) { r[i] = (__bf16)lo[i]; r[i + 8] = (__bf16)hi[i]; }
  return r;
}
// A-fragment from f32 matrix + bf16 matrix (sum in f32, round to bf16)
__device__ __forceinline__ bf16x16 afrag_mix(const float* p, const __bf16* q) {
  f32x8  lo = *(const f32x8*)(p),  hi = *(const f32x8*)(p + 16);
  bf16x8 lq = *(const bf16x8*)(q), hq = *(const bf16x8*)(q + 16);
  bf16x16 r;
#pragma unroll
  for (int i = 0; i < 8; ++i) {
    r[i]     = (__bf16)(lo[i] + (float)lq[i]);
    r[i + 8] = (__bf16)(hi[i] + (float)hq[i]);
  }
  return r;
}
// B-fragment (32x16 bf16): lane holds col lane%16; K = {0..7,16..23} (lanes<16)
// or {8..15,24..31} (lanes>=16). Weights stored transposed [N][K] row-major.
__device__ __forceinline__ bf16x16 bfrag_bf16(const __bf16* p) {
  bf16x8 lo = *(const bf16x8*)(p);
  bf16x8 hi = *(const bf16x8*)(p + 16);
  return combine16(lo, hi);
}
__device__ __forceinline__ float sigm(float x) { return 1.0f / (1.0f + __expf(-x)); }

// ---------------------------------------------------------------------------
__global__ __launch_bounds__(256) void kA_zero(int* a, int* b, int n) {
  int i = blockIdx.x * 256 + threadIdx.x;
  if (i < n) { a[i] = 0; b[i] = 0; }
}

__global__ __launch_bounds__(256) void kB_count(const int* __restrict__ dst,
                                                int* __restrict__ counts) {
  int e = blockIdx.x * 256 + threadIdx.x;
  if (e < NE) atomicAdd(&counts[dst[e]], 1);
}

// single-block exclusive scan of 50000 counts -> offs[0..NN]
__global__ __launch_bounds__(1024) void kC_scan(const int* __restrict__ counts,
                                                int* __restrict__ offs) {
  __shared__ int buf[1024];
  __shared__ int carry;
  const int tid = threadIdx.x;
  if (tid == 0) carry = 0;
  __syncthreads();
  for (int base = 0; base < NN; base += 1024) {
    int idx = base + tid;
    int x = (idx < NN) ? counts[idx] : 0;
    int v = x;
    buf[tid] = v;
    __syncthreads();
    for (int d = 1; d < 1024; d <<= 1) {
      int y = (tid >= d) ? buf[tid - d] : 0;
      __syncthreads();
      v += y;
      buf[tid] = v;
      __syncthreads();
    }
    int run = carry;
    if (idx < NN) offs[idx] = run + v - x;   // exclusive
    __syncthreads();
    if (tid == 1023) carry = run + v;
    __syncthreads();
  }
  if (tid == 0) offs[NN] = carry;
}

__global__ __launch_bounds__(256) void kD_fill(const int* __restrict__ src,
                                               const int* __restrict__ dst,
                                               const int* __restrict__ offs,
                                               int* __restrict__ cursor,
                                               int* __restrict__ csr) {
  int e = blockIdx.x * 256 + threadIdx.x;
  if (e < NE) {
    int d = dst[e];
    int p = atomicAdd(&cursor[d], 1);
    csr[offs[d] + p] = src[e];
  }
}

// pack all weights transposed to bf16 [N][K] row-major
__global__ __launch_bounds__(256) void k2_pack(const float* __restrict__ w_ih,
                                               const float* __restrict__ w_hh,
                                               const float* __restrict__ W_self,
                                               const float* __restrict__ W_neigh,
                                               const float* __restrict__ W_cls,
                                               __bf16* __restrict__ Wl,   // [512][256]
                                               __bf16* __restrict__ Wsg,  // [128][128]
                                               __bf16* __restrict__ Wc) { // [ 64][128]
  const int NL = 512 * 256, NS = 128 * 128, NC = 64 * 128;
  int idx = blockIdx.x * 256 + threadIdx.x;
  if (idx < NL) {
    int n = idx >> 8, k = idx & 255;
    float v = (k < HID) ? w_ih[(size_t)k * (4 * HID) + n]
                        : w_hh[(size_t)(k - HID) * (4 * HID) + n];
    Wl[idx] = (__bf16)v;
  } else if (idx < NL + NS) {
    int i = idx - NL, n = i >> 7, k = i & 127;
    float v = (k < INDIM) ? W_self[(size_t)k * HID + n]
                          : W_neigh[(size_t)(k - INDIM) * HID + n];
    Wsg[i] = (__bf16)v;
  } else if (idx < NL + NS + NC) {
    int i = idx - NL - NS, n = i >> 7, k = i & 127;
    Wc[i] = (__bf16)W_cls[(size_t)k * OUTDIM + n];
  }
}

// wave-per-node: gather-sum in-neighbor rows (scalar CSR walk, coalesced
// 256B feature loads, L2-resident), write A_sage = bf16([x | sum/max(deg,1)])
__global__ __launch_bounds__(256) void kE_asage(const float* __restrict__ nf,
                                                const int* __restrict__ offs,
                                                const int* __restrict__ csr,
                                                __bf16* __restrict__ A) {
  int gid = blockIdx.x * 256 + threadIdx.x;
  int n = __builtin_amdgcn_readfirstlane(gid >> 5);  // wave-uniform node id
  if (n >= NN) return;
  const int lane = threadIdx.x & 31;
  const int beg = offs[n], end = offs[n + 1];        // scalar loads
  float a0 = 0.0f, a1 = 0.0f;
  for (int i = beg; i < end; ++i) {
    int s = csr[i];                                  // scalar load (KMcnt)
    a0 += nf[(size_t)s * INDIM + lane];
    a1 += nf[(size_t)s * INDIM + 32 + lane];
  }
  const float inv = 1.0f / fmaxf((float)(end - beg), 1.0f);
  __bf16* row = A + (size_t)n * HID;
  row[lane]      = (__bf16)nf[(size_t)n * INDIM + lane];
  row[32 + lane] = (__bf16)nf[(size_t)n * INDIM + 32 + lane];
  row[64 + lane] = (__bf16)(a0 * inv);
  row[96 + lane] = (__bf16)(a1 * inv);
}

// h_struct[NPAD,128] = A_sage[NPAD,128] @ W(128,128) + b ; 128 rows/block
__global__ __launch_bounds__(256) void k4_sage_gemm(const __bf16* __restrict__ A,
                                                    const __bf16* __restrict__ Wt,
                                                    const float* __restrict__ bias,
                                                    float* __restrict__ Hs) {
  const int wave = threadIdx.x >> 5, lane = threadIdx.x & 31;
  const int mlan = lane & 15, half = lane >> 4;
  const int khA = half << 3, khB = half << 3;
  const int arow = blockIdx.x * 128 + (wave << 4) + mlan;
  bf16x16 af[4];
  const __bf16* ap = A + (size_t)arow * HID + khA;   // rows padded: no guard
#pragma unroll
  for (int kt = 0; kt < 4; ++kt) af[kt] = afrag_bf16(ap + kt * 32);
  const int mrow = blockIdx.x * 128 + (wave << 4) + (half << 3);  // r=0 row
#pragma unroll
  for (int j = 0; j < 8; ++j) {
    f32x8 acc = zacc();                              // -> inline-0 C operand
#pragma unroll
    for (int kt = 0; kt < 4; ++kt) {
      bf16x16 bf = bfrag_bf16(Wt + (size_t)((j << 4) + mlan) * HID + kt * 32 + khB);
      acc = __builtin_amdgcn_wmma_f32_16x16x32_bf16(false, af[kt], false, bf,
                                                    (short)0, acc, false, false);
    }
    const float bb = bias[(j << 4) + mlan];
    float* op = Hs + (size_t)mrow * HID + (j << 4) + mlan;
#pragma unroll
    for (int r = 0; r < 8; ++r) op[(size_t)r * HID] = acc[r] + bb;
  }
}

// Fused 8-step LSTM. 128 nodes/block, weights bf16 resident in LDS (256KB),
// h bf16 in LDS (32KB), c in VGPRs. Each wave owns 16 node-rows => the
// recurrent h read/write never crosses waves => no barriers in the t-loop.
__global__ __launch_bounds__(256) void k5_lstm(const float* __restrict__ hist,
                                               const float* __restrict__ bl,
                                               const __bf16* __restrict__ gW, // [512][256]
                                               __bf16* __restrict__ hT) {    // [NPAD][128]
  extern __shared__ char smem[];
  __bf16* sW = (__bf16*)smem;                        // 512*256 = 256KB
  __bf16* sH = (__bf16*)(smem + 512 * 256 * 2);      // 128*128 =  32KB

  const int tid = threadIdx.x;
  {  // stage weights (L2-resident) into LDS; zero h
    const bf16x8* s8 = (const bf16x8*)gW;
    bf16x8* d8 = (bf16x8*)sW;
    for (int i = tid; i < 512 * 256 / 8; i += 256) d8[i] = s8[i];
    bf16x8 z;
#pragma unroll
    for (int i = 0; i < 8; ++i) z[i] = (__bf16)0.0f;
    for (int i = tid; i < 128 * 128 / 8; i += 256) ((bf16x8*)sH)[i] = z;
  }
  __syncthreads();

  const int wave = tid >> 5, lane = tid & 31;
  const int mlan = lane & 15, half = lane >> 4;
  const int khA = half << 3, khB = half << 3;
  const int row0 = wave << 4;                 // node-local
  const int gnode = blockIdx.x * 128 + row0 + mlan;
  const bool ok = gnode < NN;                 // hist is unpadded input: guard

  float bv[4][8];                             // per-lane column biases, hoisted
#pragma unroll
  for (int g = 0; g < 4; ++g)
#pragma unroll
    for (int j = 0; j < 8; ++j) bv[g][j] = bl[g * HID + (j << 4) + mlan];

  f32x8 c_reg[8];
#pragma unroll
  for (int j = 0; j < 8; ++j) c_reg[j] = zacc();

#pragma unroll 1
  for (int t = 0; t < THIST; ++t) {
    // A = [x_t | h]  (K = 256), 8 fragments of 16x32
    bf16x16 af[8];
    const float* xr = hist + (size_t)gnode * (THIST * HID) + t * HID + khA;
#pragma unroll
    for (int kt = 0; kt < 4; ++kt) af[kt] = ok ? afrag_f32(xr + kt * 32) : zfrag();
    const __bf16* hr = sH + (size_t)(row0 + mlan) * HID + khA;
#pragma unroll
    for (int kt = 0; kt < 4; ++kt) af[4 + kt] = afrag_bf16(hr + kt * 32);

#pragma unroll
    for (int j = 0; j < 8; ++j) {
      f32x8 acc[4];
#pragma unroll
      for (int g = 0; g < 4; ++g) acc[g] = zacc();   // -> inline-0 C operand
#pragma unroll
      for (int kt = 0; kt < 8; ++kt) {
#pragma unroll
        for (int g = 0; g < 4; ++g) {
          bf16x16 bf = bfrag_bf16(sW + (size_t)(g * HID + (j << 4) + mlan) * 256 +
                                  kt * 32 + khB);
          acc[g] = __builtin_amdgcn_wmma_f32_16x16x32_bf16(
              false, af[kt], false, bf, (short)0, acc[g], false, false);
        }
      }
      // fused gate nonlinearities on the C/D fragment layout (bias added here)
#pragma unroll
      for (int r = 0; r < 8; ++r) {
        float iv = sigm(acc[0][r] + bv[0][j]);
        float fv = sigm(acc[1][r] + bv[1][j]);
        float gv = tanhf(acc[2][r] + bv[2][j]);
        float ov = sigm(acc[3][r] + bv[3][j]);
        float c = fv * c_reg[j][r] + iv * gv;
        c_reg[j][r] = c;
        float h = ov * tanhf(c);
        int mm = row0 + r + (half << 3);       // node-local row of this element
        sH[(size_t)mm * HID + (j << 4) + mlan] = (__bf16)h; // same-wave, in-order
      }
    }
  }

  // write h_T (bf16) for this wave's 16 rows; hT padded => branch-free b128s
  {
    const bf16x8* s8 = (const bf16x8*)(sH + (size_t)row0 * HID);
    bf16x8* d8 = (bf16x8*)(hT + ((size_t)blockIdx.x * 128 + row0) * HID);
#pragma unroll
    for (int i = 0; i < 8; ++i) d8[lane + 32 * i] = s8[lane + 32 * i];
  }
}

// out[N,64] = bf16(h_struct + h_T) @ W_cls(128,64) + b
__global__ __launch_bounds__(256) void k6_cls(const float* __restrict__ Hs,
                                              const __bf16* __restrict__ Ht,
                                              const __bf16* __restrict__ Wt, // [64][128]
                                              const float* __restrict__ bias,
                                              float* __restrict__ out) {
  const int wave = threadIdx.x >> 5, lane = threadIdx.x & 31;
  const int mlan = lane & 15, half = lane >> 4;
  const int khA = half << 3, khB = half << 3;
  const int arow = blockIdx.x * 128 + (wave << 4) + mlan;
  bf16x16 af[4];
  const size_t base = (size_t)arow * HID + khA;      // padded: no guard
#pragma unroll
  for (int kt = 0; kt < 4; ++kt)
    af[kt] = afrag_mix(Hs + base + kt * 32, Ht + base + kt * 32);
  const int mrow = blockIdx.x * 128 + (wave << 4) + (half << 3);  // r=0 row
  const bool full = (blockIdx.x + 1) * 128 <= NN;    // uniform branch
#pragma unroll
  for (int j = 0; j < 4; ++j) {
    f32x8 acc = zacc();                              // -> inline-0 C operand
#pragma unroll
    for (int kt = 0; kt < 4; ++kt) {
      bf16x16 bf = bfrag_bf16(Wt + (size_t)((j << 4) + mlan) * HID + kt * 32 + khB);
      acc = __builtin_amdgcn_wmma_f32_16x16x32_bf16(false, af[kt], false, bf,
                                                    (short)0, acc, false, false);
    }
    const float bb = bias[(j << 4) + mlan];
    float* op = out + (size_t)mrow * OUTDIM + (j << 4) + mlan;
    if (full) {
#pragma unroll
      for (int r = 0; r < 8; ++r) op[(size_t)r * OUTDIM] = acc[r] + bb;
    } else {
#pragma unroll
      for (int r = 0; r < 8; ++r)
        if (mrow + r < NN) op[(size_t)r * OUTDIM] = acc[r] + bb;
    }
  }
}

// ---------------------------------------------------------------------------
extern "C" void kernel_launch(void* const* d_in, const int* in_sizes, int n_in,
                              void* d_out, int out_size, void* d_ws, size_t ws_size,
                              hipStream_t stream) {
  const float* node_feats = (const float*)d_in[0];
  const float* hist       = (const float*)d_in[1];
  const int*   src        = (const int*)  d_in[2];
  const int*   dst        = (const int*)  d_in[3];
  const float* W_self     = (const float*)d_in[4];
  const float* W_neigh    = (const float*)d_in[5];
  const float* b_sage     = (const float*)d_in[6];
  const float* w_ih       = (const float*)d_in[7];
  const float* w_hh       = (const float*)d_in[8];
  const float* b_lstm     = (const float*)d_in[9];
  const float* W_cls      = (const float*)d_in[10];
  const float* b_cls      = (const float*)d_in[11];
  float* out = (float*)d_out;

  // workspace layout (~55.4 MB), 256B-aligned, matrix rows padded to NPAD
  char* ws = (char*)d_ws;
  int*    counts   = (int*)   (ws + 0);           // 50000*4     =    200,000
  int*    cursor   = (int*)   (ws + 200192);      // 50000*4     =    200,000
  int*    offs     = (int*)   (ws + 400640);      // 50001*4     =    200,004
  int*    csr      = (int*)   (ws + 600832);      // 800000*4    =  3,200,000
  float*  h_struct = (float*) (ws + 3800832);     // NPAD*128*4  = 25,624,576
  __bf16* hT       = (__bf16*)(ws + 29425408);    // NPAD*128*2  = 12,812,288
  __bf16* A_sage   = (__bf16*)(ws + 42237696);    // NPAD*128*2  = 12,812,288
  __bf16* Wl       = (__bf16*)(ws + 55049984);    // 512*256*2   =    262,144
  __bf16* Wsg      = (__bf16*)(ws + 55312128);    // 128*128*2   =     32,768
  __bf16* Wc       = (__bf16*)(ws + 55344896);    // 64*128*2    =     16,384

  kA_zero<<<(NN + 255) / 256, 256, 0, stream>>>(counts, cursor, NN);
  kB_count<<<(NE + 255) / 256, 256, 0, stream>>>(dst, counts);
  kC_scan<<<1, 1024, 0, stream>>>(counts, offs);
  kD_fill<<<(NE + 255) / 256, 256, 0, stream>>>(src, dst, offs, cursor, csr);

  const int npack = 512 * 256 + 128 * 128 + 64 * 128;
  k2_pack<<<(npack + 255) / 256, 256, 0, stream>>>(w_ih, w_hh, W_self, W_neigh,
                                                   W_cls, Wl, Wsg, Wc);

  kE_asage<<<(int)(((long)NN * 32 + 255) / 256), 256, 0, stream>>>(
      node_feats, offs, csr, A_sage);

  const int nblk = NPAD / 128;  // 391
  k4_sage_gemm<<<nblk, 256, 0, stream>>>(A_sage, Wsg, b_sage, h_struct);

  const size_t smem_lstm = (size_t)(512 * 256 + 128 * 128) * sizeof(__bf16); // 288KB
  k5_lstm<<<nblk, 256, smem_lstm, stream>>>(hist, b_lstm, Wl, hT);

  k6_cls<<<nblk, 256, 0, stream>>>(h_struct, hT, Wc, b_cls, out);
}
